// Dlrm_72181220376948
// MI455X (gfx1250) — matrix operand order repaired and checked by reading, not compiled, not run on hardware
//
#include <hip/hip_runtime.h>
#include <stdint.h>

#define WB 32768
#define NVOC 100000
#define NTAB 26

typedef __bf16 v16bf __attribute__((ext_vector_type(16)));
typedef float  v8f   __attribute__((ext_vector_type(8)));

union ABfrag { v16bf v; uint32_t u[8]; };
union Cfrag  { v8f v; float f[8]; };

static __device__ __forceinline__ uint16_t f32_to_bf16(float x) {
    union { float f; uint32_t u; } c; c.f = x;
    uint32_t u = c.u;
    uint32_t r = u + 0x7FFFu + ((u >> 16) & 1u);   // round-to-nearest-even
    return (uint16_t)(r >> 16);
}

// CDNA5 async copy: global -> LDS, 16B per lane, tracked by ASYNCcnt.
// vdst = per-lane LDS byte address, vaddr = per-lane 64-bit global address.
static __device__ __forceinline__ void async_ld_b128(uint32_t lds_off,
                                                     const void* gaddr) {
    asm volatile("global_load_async_to_lds_b128 %0, %1, off"
                 :: "v"(lds_off), "v"((uint64_t)(uintptr_t)gaddr)
                 : "memory");
}
static __device__ __forceinline__ void wait_async0() {
    asm volatile("s_wait_asynccnt 0" ::: "memory");
}

// ---------------------------------------------------------------------------
// Weight conversion: W (Kreal x N, f32 row-major) -> Wt (Npad x Kpad, bf16),
// transposed + zero padded so B-fragment loads are packed pairs per lane.
// ---------------------------------------------------------------------------
__global__ __launch_bounds__(256) void convert_wt_kernel(
    const float* __restrict__ W, uint16_t* __restrict__ Wt,
    int Kreal, int Kpad, int N, int Npad)
{
    int idx = blockIdx.x * 256 + threadIdx.x;
    int total = Npad * Kpad;
    if (idx >= total) return;
    int n = idx / Kpad;
    int k = idx - n * Kpad;
    float v = (n < N && k < Kreal) ? W[(size_t)k * N + n] : 0.0f;
    Wt[idx] = f32_to_bf16(v);
}

// Pad numerical (B x 13 f32) -> x0 (B x 32 bf16, zeros beyond col 12)
__global__ __launch_bounds__(256) void pad_input_kernel(
    const float* __restrict__ num, uint16_t* __restrict__ x0)
{
    int idx = blockIdx.x * 256 + threadIdx.x;      // over B*32
    int row = idx >> 5;
    int c = idx & 31;
    float v = (c < 13) ? num[(size_t)row * 13 + c] : 0.0f;
    x0[idx] = f32_to_bf16(v);
}

// ---------------------------------------------------------------------------
// bf16 WMMA GEMM: out = act(A[B x K] * Wt^T + bias)
//   A: bf16 row-major (stride K), Wt: bf16 (Npad x K) transposed weights.
//   Block: 256 threads = 8 waves; block tile 128(M) x 64(N).
//   Wave tile 32x32: 4 accumulators, 4 x v_wmma_f32_16x16x32_bf16 per k-step,
//   2-way register reuse of both a- and b-fragments.
//   K-slabs double-buffered in LDS (stride 56 elem = 112B: 16B-aligned b128,
//   28-dword row phase -> conflict-free 16-lane fragment reads), filled by
//   global_load_async_to_lds_b128 (ASYNCcnt) overlapped with the WMMA issue.
//   MODE 0: bf16 output + ReLU (N multiple of 64). MODE 1: f32 col-0 output.
// ---------------------------------------------------------------------------
#define LDA 56
template <int MODE>
__global__ __launch_bounds__(256) void gemm_bf16_kernel(
    const uint16_t* __restrict__ A, const uint16_t* __restrict__ Wt,
    const float* __restrict__ bias,
    uint16_t* __restrict__ outB, float* __restrict__ outF,
    int K, int Nstride, int Nvalid)
{
    __shared__ uint16_t As[2][128 * LDA];   // ping-pong 128 x 32 slabs
    __shared__ uint16_t Ws[2][ 64 * LDA];   // ping-pong  64 x 32 slabs

    const int tid  = threadIdx.x;
    const int lane = tid & 31;
    const int wave = tid >> 5;
    const int half = lane >> 4;
    const int l16  = lane & 15;
    const int m0   = blockIdx.x * 128;
    const int n0   = blockIdx.y * 64;
    const int wm   = (wave & 3) * 32;    // wave M offset within block tile
    const int wn   = (wave >> 2) * 32;   // wave N offset within block tile

    // cooperative-load geometry: 16B chunks (8 bf16)
    const int arow0 = tid >> 2,        acol = (tid & 3) * 8;       // A chunk 0
    const int arow1 = (tid + 256) >> 2;                            // A chunk 1
    const int wrow  = tid >> 2;                                    // W chunk

    Cfrag acc[2][2];
    acc[0][0] = Cfrag{}; acc[0][1] = Cfrag{};
    acc[1][0] = Cfrag{}; acc[1][1] = Cfrag{};

    auto issue_slab = [&](int kk, int pp) {
        async_ld_b128((uint32_t)(uintptr_t)&As[pp][arow0 * LDA + acol],
                      &A [(size_t)(m0 + arow0) * K + kk + acol]);
        async_ld_b128((uint32_t)(uintptr_t)&As[pp][arow1 * LDA + acol],
                      &A [(size_t)(m0 + arow1) * K + kk + acol]);
        async_ld_b128((uint32_t)(uintptr_t)&Ws[pp][wrow  * LDA + acol],
                      &Wt[(size_t)(n0 + wrow ) * K + kk + acol]);
    };

    // prologue: fill buffer 0
    issue_slab(0, 0);
    wait_async0();
    __syncthreads();

    int p = 0;
    for (int k0 = 0; k0 < K; k0 += 32) {
        // kick off next slab into the other buffer (overlaps with WMMAs)
        if (k0 + 32 < K) issue_slab(k0 + 32, p ^ 1);

        ABfrag a[2], b[2];
#pragma unroll
        for (int t = 0; t < 2; ++t) {
            const int ra = wm + t * 16 + l16;
            const int rb = wn + t * 16 + l16;
#pragma unroll
            for (int i = 0; i < 8; ++i) {
                int ka = ((i & 4) ? 16 : 0) + 2 * (i & 3) + half * 8;
                a[t].u[i] = *(const uint32_t*)&As[p][ra * LDA + ka];
                int kb = ((i & 4) ? 16 : 0) + 2 * (i & 3) + half * 8;
                b[t].u[i] = *(const uint32_t*)&Ws[p][rb * LDA + kb];
            }
        }
#pragma unroll
        for (int mi = 0; mi < 2; ++mi)
#pragma unroll
            for (int ni = 0; ni < 2; ++ni)
                acc[mi][ni].v = __builtin_amdgcn_wmma_f32_16x16x32_bf16(
                    false, a[mi].v, false, b[ni].v, (short)0,
                    acc[mi][ni].v, false, false);

        if (k0 + 32 < K) wait_async0();   // next buffer fully in LDS
        __syncthreads();                  // ...and visible to all waves
        p ^= 1;
    }

    // epilogue: bias + activation + store
#pragma unroll
    for (int ni = 0; ni < 2; ++ni) {
        const int n = n0 + wn + ni * 16 + l16;
        const float bs = (n < Nvalid) ? bias[n] : 0.0f;
#pragma unroll
        for (int mi = 0; mi < 2; ++mi) {
#pragma unroll
            for (int r = 0; r < 8; ++r) {
                int m = m0 + wm + mi * 16 + r + half * 8;
                float v = acc[mi][ni].f[r] + bs;
                if (MODE == 0) {
                    v = v > 0.0f ? v : 0.0f;
                    outB[(size_t)m * Nstride + n] = f32_to_bf16(v);
                } else {
                    if (n == 0) outF[m] = v;   // single real output column
                }
            }
        }
    }
}

// ---------------------------------------------------------------------------
// Fused embedding-gather + pairwise interaction.
// One block (128 thr = 4 waves) per batch row. feats (27x128, pad to 32) in
// LDS as bf16; Z = F*F^T via 4 wave-tiles of 16x16, K-loop of 4 x WMMA.
// Emits top-MLP input row: [bottom(128) | triu pairs(351) | zero(1)] bf16.
// ---------------------------------------------------------------------------
__global__ __launch_bounds__(128) void interact_kernel(
    const uint16_t* __restrict__ bot,   // B x 128 bf16
    const int* __restrict__ cat,        // B x 26
    const float* __restrict__ emb,      // 26 x V x 128 f32
    uint16_t* __restrict__ topin)       // B x 480 bf16
{
    __shared__ uint16_t F[32][136];
    const int row = blockIdx.x;
    const int tid = threadIdx.x;        // 0..127

    uint16_t bv = bot[(size_t)row * 128 + tid];
    F[0][tid] = bv;
    topin[(size_t)row * 480 + tid] = bv;

#pragma unroll 1
    for (int t = 0; t < NTAB; ++t) {
        int idx = cat[(size_t)row * NTAB + t];
        const float* p = emb + ((size_t)t * NVOC + (size_t)idx) * 128;
        F[1 + t][tid] = f32_to_bf16(p[tid]);
    }
#pragma unroll
    for (int rr = 27; rr < 32; ++rr) F[rr][tid] = 0;
    if (tid == 0) topin[(size_t)row * 480 + 479] = 0;
    __syncthreads();

    const int lane = tid & 31;
    const int wave = tid >> 5;
    const int half = lane >> 4;
    const int l16  = lane & 15;
    const int mt = (wave >> 1) * 16;
    const int nt = (wave & 1) * 16;

    Cfrag acc{};
#pragma unroll
    for (int k0 = 0; k0 < 128; k0 += 32) {
        ABfrag a, b;
#pragma unroll
        for (int i = 0; i < 8; ++i) {
            int ka = ((i & 4) ? 16 : 0) + 2 * (i & 3) + half * 8;
            a.u[i] = *(const uint32_t*)&F[mt + l16][k0 + ka];
            int kb = ((i & 4) ? 16 : 0) + 2 * (i & 3) + half * 8;
            b.u[i] = *(const uint32_t*)&F[nt + l16][k0 + kb];
        }
        acc.v = __builtin_amdgcn_wmma_f32_16x16x32_bf16(
            false, a.v, false, b.v, (short)0, acc.v, false, false);
    }

#pragma unroll
    for (int r = 0; r < 8; ++r) {
        int m = mt + r + half * 8;
        int n = nt + l16;
        if (m < n && n < 27) {
            int p = m * (53 - m) / 2 + (n - m - 1);   // flat triu(k=1) index
            topin[(size_t)row * 480 + 128 + p] = f32_to_bf16(acc.f[r]);
        }
    }
}

// ---------------------------------------------------------------------------
extern "C" void kernel_launch(void* const* d_in, const int* in_sizes, int n_in,
                              void* d_out, int out_size, void* d_ws, size_t ws_size,
                              hipStream_t stream)
{
    (void)in_sizes; (void)n_in; (void)out_size; (void)ws_size;

    const float* numerical = (const float*)d_in[0];
    const int*   cat       = (const int*)d_in[1];
    const float* emb       = (const float*)d_in[2];
    const float* W[8]  = { (const float*)d_in[3],  (const float*)d_in[5],
                           (const float*)d_in[7],  (const float*)d_in[9],
                           (const float*)d_in[11], (const float*)d_in[13],
                           (const float*)d_in[15], (const float*)d_in[17] };
    const float* Bv[8] = { (const float*)d_in[4],  (const float*)d_in[6],
                           (const float*)d_in[8],  (const float*)d_in[10],
                           (const float*)d_in[12], (const float*)d_in[14],
                           (const float*)d_in[16], (const float*)d_in[18] };

    const int Kr[8] = { 16, 512, 256, 480, 1024, 1024, 512, 256 };
    const int Kp[8] = { 32, 512, 256, 480, 1024, 1024, 512, 256 };
    const int Nn[8] = { 512, 256, 128, 1024, 1024, 512, 256, 1 };
    const int Np[8] = { 512, 256, 128, 1024, 1024, 512, 256, 64 };

    char* ws = (char*)d_ws;
    uint16_t* bufA   = (uint16_t*)ws;                                  // B*1024 bf16
    uint16_t* bufB   = (uint16_t*)(ws + (size_t)WB * 1024 * 2);        // B*1024 bf16
    uint16_t* bufBot = (uint16_t*)(ws + 2 * (size_t)WB * 1024 * 2);    // B*128 bf16
    size_t wo = 2 * (size_t)WB * 1024 * 2 + (size_t)WB * 128 * 2;
    uint16_t* Wt[8];
    for (int i = 0; i < 8; ++i) {
        Wt[i] = (uint16_t*)(ws + wo);
        wo += (size_t)Np[i] * Kp[i] * 2;
    }

    // weight conversion (small; L2-resident afterwards)
    for (int i = 0; i < 8; ++i) {
        int total = Np[i] * Kp[i];
        convert_wt_kernel<<<(total + 255) / 256, 256, 0, stream>>>(
            W[i], Wt[i], Kr[i], Kp[i], Nn[i], Np[i]);
    }
    pad_input_kernel<<<(WB * 32) / 256, 256, 0, stream>>>(numerical, bufA);

    uint16_t* x0 = bufA;      // B x 32
    uint16_t* x1 = bufB;      // B x 512
    uint16_t* x2 = bufA;      // B x 256   (x0 dead)
    uint16_t* bo = bufBot;    // B x 128
    uint16_t* ti = bufA;      // B x 480   (x2 dead)
    uint16_t* t1 = bufB;      // B x 1024  (x1 dead)
    uint16_t* t2 = bufA;      // B x 1024  (ti dead)
    uint16_t* t3 = bufB;      // B x 512
    uint16_t* t4 = bufA;      // B x 256
    float*    of = (float*)d_out;

    dim3 blk(256);
    // bottom MLP
    gemm_bf16_kernel<0><<<dim3(WB / 128,  512 / 64), blk, 0, stream>>>(x0, Wt[0], Bv[0], x1, nullptr,   32,  512,  512);
    gemm_bf16_kernel<0><<<dim3(WB / 128,  256 / 64), blk, 0, stream>>>(x1, Wt[1], Bv[1], x2, nullptr,  512,  256,  256);
    gemm_bf16_kernel<0><<<dim3(WB / 128,  128 / 64), blk, 0, stream>>>(x2, Wt[2], Bv[2], bo, nullptr,  256,  128,  128);
    // gather + interaction + top-input assembly
    interact_kernel<<<WB, 128, 0, stream>>>(bo, cat, emb, ti);
    // top MLP
    gemm_bf16_kernel<0><<<dim3(WB / 128, 1024 / 64), blk, 0, stream>>>(ti, Wt[3], Bv[3], t1, nullptr,  480, 1024, 1024);
    gemm_bf16_kernel<0><<<dim3(WB / 128, 1024 / 64), blk, 0, stream>>>(t1, Wt[4], Bv[4], t2, nullptr, 1024, 1024, 1024);
    gemm_bf16_kernel<0><<<dim3(WB / 128,  512 / 64), blk, 0, stream>>>(t2, Wt[5], Bv[5], t3, nullptr, 1024,  512,  512);
    gemm_bf16_kernel<0><<<dim3(WB / 128,  256 / 64), blk, 0, stream>>>(t3, Wt[6], Bv[6], t4, nullptr,  512,  256,  256);
    gemm_bf16_kernel<1><<<dim3(WB / 128,          1), blk, 0, stream>>>(t4, Wt[7], Bv[7], nullptr, of,  256,    1,    1);
}
